// ConvLayers_23605140259364
// MI455X (gfx1250) — matrix operand order repaired and checked by reading, not compiled, not run on hardware
//
#include <hip/hip_runtime.h>
#include <hip/hip_bf16.h>

typedef __attribute__((ext_vector_type(16))) _Float16 v16h;
typedef __attribute__((ext_vector_type(8)))  float    v8f;

#define N_NODES  50000
#define N_EDGES  1600000
#define D_IN     64
#define D_EDGE   64
#define D_OUT    128
#define N_GRAPHS 64

// ---------------- utility ----------------
__global__ void zero_f32(float* __restrict__ p, long n) {
  long i = (long)blockIdx.x * blockDim.x + threadIdx.x;
  if (i < n) p[i] = 0.0f;
}

// ---------------- once-per-launch: T = segment_sum(edge_attr, dst), deg = in-degree ----------------
__global__ void edge_attr_agg(const float* __restrict__ ea, const int* __restrict__ ei,
                              float* __restrict__ T, float* __restrict__ deg) {
  const int chunks = D_EDGE / 4; // 16
  long i = (long)blockIdx.x * blockDim.x + threadIdx.x;
  if (i >= (long)N_EDGES * chunks) return;
  int e = (int)(i / chunks);
  int c = (int)(i % chunks) * 4;
  int dst = ei[N_EDGES + e];
  const float4 v = *(const float4*)(ea + (size_t)e * D_EDGE + c);
  float* o = T + (size_t)dst * D_EDGE + c;
  atomicAdd(o + 0, v.x);
  atomicAdd(o + 1, v.y);
  atomicAdd(o + 2, v.z);
  atomicAdd(o + 3, v.w);
  if (c == 0) atomicAdd(deg + dst, 1.0f);
}

// ---------------- per-layer: S[dst] += x[src]  (x, S, edge_index all L2-resident) ----------------
__global__ void scatter_feat(const float* __restrict__ x, const int* __restrict__ ei,
                             float* __restrict__ S, int in_dim) {
  const int chunks = in_dim >> 2;
  long i = (long)blockIdx.x * blockDim.x + threadIdx.x;
  if (i >= (long)N_EDGES * chunks) return;
  int e = (int)(i / chunks);
  int c = (int)(i % chunks) * 4;
  int src = ei[e];
  int dst = ei[N_EDGES + e];
  const float4 v = *(const float4*)(x + (size_t)src * in_dim + c);
  float* o = S + (size_t)dst * in_dim + c;
  atomicAdd(o + 0, v.x);
  atomicAdd(o + 1, v.y);
  atomicAdd(o + 2, v.z);
  atomicAdd(o + 3, v.w);
}

// ---------------- pack node features [x | S | T] -> f16 in WMMA A-fragment layout ----------------
// Apk is organized as blocks of 512 halves per (mt, c):
//   Apk[((mt*(Kc/32) + c)*32 + lane)*16 + j]  ==  A(m = mt*16 + lane%16, k = c*32 + kcw)
// with hs = lane/16 and kcw <-> (j,hs) per the ISA 16-bit A layout:
//   hs=0: kcw = (j<8) ? j   : j+8
//   hs=1: kcw = (j<8) ? j+8 : j+16
__global__ void pack_feat(const float* __restrict__ x, const float* __restrict__ S,
                          const float* __restrict__ T, _Float16* __restrict__ Apk, int in_dim) {
  const int Kc = 2 * in_dim + D_EDGE;
  long i = (long)blockIdx.x * blockDim.x + threadIdx.x;
  if (i >= (long)N_NODES * Kc) return;
  int blk  = (int)(i >> 9);        // (mt, c) block
  int t    = (int)(i & 511);
  int lane = t >> 4;
  int j    = t & 15;
  const int NC = Kc >> 5;
  int mt = blk / NC;
  int c  = blk % NC;
  int hs = lane >> 4;
  int m  = (mt << 4) + (lane & 15);
  int kcw = (hs == 0) ? ((j < 8) ? j : j + 8) : ((j < 8) ? j + 8 : j + 16);
  int k = (c << 5) + kcw;
  float v;
  if (k < in_dim)            v = x[(size_t)m * in_dim + k];
  else if (k < 2 * in_dim)   v = S[(size_t)m * in_dim + (k - in_dim)];
  else                       v = T[(size_t)m * D_EDGE + (k - 2 * in_dim)];
  Apk[i] = (_Float16)v;
}

// ---------------- pack stacked weights [Wr ; We] -> f16 in WMMA B-fragment layout ----------------
// Bpk blocks of 512 halves per (c, nt):
//   Bpk[((c*8 + nt)*32 + lane)*16 + j] == W(k = c*32 + ((j<8) ? j : j+8) + 8*(lane/16), n = nt*16 + lane%16)
__global__ void pack_w(const float* __restrict__ Wr, const float* __restrict__ We,
                       _Float16* __restrict__ Bpk, int in_dim) {
  const int Kc = 2 * in_dim + D_EDGE;
  int i = blockIdx.x * blockDim.x + threadIdx.x;
  if (i >= Kc * D_OUT) return;
  int blk  = i >> 9;               // (c, nt)
  int t    = i & 511;
  int lane = t >> 4;
  int j    = t & 15;
  int c  = blk >> 3;
  int nt = blk & 7;
  int n  = (nt << 4) + (lane & 15);
  int k  = (c << 5) + ((j < 8) ? j : j + 8) + 8 * (lane >> 4);
  float v = (k < in_dim) ? Wr[k * D_OUT + n] : We[(k - in_dim) * D_OUT + n];
  Bpk[i] = (_Float16)v;
}

// ---------------- node-level GEMM via v_wmma_f32_16x16x32_f16 ----------------
// One wave (block of 32) computes a 16(M) x 128(N) strip: 8 accumulators, A amortized
// across 8 back-to-back WMMAs per K-chunk. KC compile-time -> fully unrolled.
// All waves full (50000 = 3125*16) -> EXEC all-ones everywhere.
template <int KC>
__global__ void __launch_bounds__(32) gemm_node(const _Float16* __restrict__ Apk,
                                                const _Float16* __restrict__ Bpk,
                                                const float* __restrict__ deg,
                                                const float* __restrict__ be,
                                                const float* __restrict__ br,
                                                float* __restrict__ out) {
  constexpr int NC = KC / 32;
  const int lane = threadIdx.x;      // 0..31
  const int mt   = blockIdx.x;       // 0..3124

  const v16h* A = (const v16h*)Apk + (size_t)mt * NC * 32 + lane;
  const v16h* B = (const v16h*)Bpk + lane;

  v8f acc[8] = {};
#pragma unroll
  for (int c = 0; c < NC; ++c) {
    v16h a = A[c * 32];
#pragma unroll
    for (int nt = 0; nt < 8; ++nt) {
      v16h b = B[(c * 8 + nt) * 32];
      acc[nt] = __builtin_amdgcn_wmma_f32_16x16x32_f16(
          /*neg_a=*/false, a, /*neg_b=*/false, b,
          /*c_mod=*/(short)0, acc[nt], /*reuse_a=*/false, /*reuse_b=*/false);
    }
  }

  // C/D f32 layout: VGPR r -> row m0 + r + 8*(lane/16), column n0 + lane%16
  const int hs = lane >> 4;
  const int lg = lane & 15;
  const int m0 = mt << 4;
  float dv[8];
#pragma unroll
  for (int r = 0; r < 8; ++r) dv[r] = deg[m0 + r + 8 * hs];   // contiguous -> b128 pairs
#pragma unroll
  for (int nt = 0; nt < 8; ++nt) {
    const int n = (nt << 4) + lg;
    const float bev = be[n];
    const float brv = br[n];
#pragma unroll
    for (int r = 0; r < 8; ++r) {
      const int mm = m0 + r + 8 * hs;
      float v = acc[nt][r] + brv + dv[r] * bev;
      out[(size_t)mm * D_OUT + n] = v > 0.0f ? v : 0.0f;
    }
  }
}

// ---------------- final global_add_pool over graph ids ----------------
__global__ void pool_add(const float* __restrict__ h, const int* __restrict__ batch,
                         float* __restrict__ out) {
  long i = (long)blockIdx.x * blockDim.x + threadIdx.x;
  if (i >= (long)N_NODES * D_OUT) return;
  int n = (int)(i / D_OUT);
  int f = (int)(i % D_OUT);
  atomicAdd(&out[(size_t)batch[n] * D_OUT + f], h[i]);
}

// ---------------- driver ----------------
extern "C" void kernel_launch(void* const* d_in, const int* in_sizes, int n_in,
                              void* d_out, int out_size, void* d_ws, size_t ws_size,
                              hipStream_t stream) {
  const float* x0    = (const float*)d_in[0];
  const int*   ei    = (const int*)d_in[1];   // [2, E]: row0 = src, row1 = dst
  const float* ea    = (const float*)d_in[2];
  const int*   batch = (const int*)d_in[3];
  const float* We[3] = {(const float*)d_in[4], (const float*)d_in[8],  (const float*)d_in[12]};
  const float* be[3] = {(const float*)d_in[5], (const float*)d_in[9],  (const float*)d_in[13]};
  const float* Wr[3] = {(const float*)d_in[6], (const float*)d_in[10], (const float*)d_in[14]};
  const float* br[3] = {(const float*)d_in[7], (const float*)d_in[11], (const float*)d_in[15]};
  float* outp = (float*)d_out;

  // workspace carve-up
  char* ws = (char*)d_ws;
  size_t off = 0;
  auto carve = [&](size_t bytes) {
    void* p = ws + off;
    off = (off + bytes + 255) & ~(size_t)255;
    return p;
  };
  float*     h0  = (float*)carve((size_t)N_NODES * D_OUT * sizeof(float));
  float*     h1  = (float*)carve((size_t)N_NODES * D_OUT * sizeof(float));
  float*     S   = (float*)carve((size_t)N_NODES * D_OUT * sizeof(float));   // max in_dim=128
  float*     T   = (float*)carve((size_t)N_NODES * D_EDGE * sizeof(float));
  float*     deg = (float*)carve((size_t)N_NODES * sizeof(float));
  _Float16*  Apk = (_Float16*)carve((size_t)N_NODES * 320 * sizeof(_Float16)); // max Kc=320
  _Float16*  Bpk = (_Float16*)carve((size_t)320 * D_OUT * sizeof(_Float16));

  const int TB = 256;
  auto nblk = [](long n, int tb) { return (unsigned)((n + tb - 1) / tb); };

  // layer-invariant: T, deg
  zero_f32<<<nblk((long)N_NODES * D_EDGE, TB), TB, 0, stream>>>(T, (long)N_NODES * D_EDGE);
  zero_f32<<<nblk(N_NODES, TB), TB, 0, stream>>>(deg, N_NODES);
  edge_attr_agg<<<nblk((long)N_EDGES * (D_EDGE / 4), TB), TB, 0, stream>>>(ea, ei, T, deg);

  const float* cur = x0;
  int in_dim = D_IN;
  float* hbuf[2] = {h0, h1};
  for (int l = 0; l < 3; ++l) {
    const int Kc = 2 * in_dim + D_EDGE; // 192, 320, 320
    // S = scatter_add(x[src], dst)
    zero_f32<<<nblk((long)N_NODES * in_dim, TB), TB, 0, stream>>>(S, (long)N_NODES * in_dim);
    scatter_feat<<<nblk((long)N_EDGES * (in_dim / 4), TB), TB, 0, stream>>>(cur, ei, S, in_dim);
    // pack [x | S | T] and [Wr ; We] into fragment-native f16 layouts
    pack_feat<<<nblk((long)N_NODES * Kc, TB), TB, 0, stream>>>(cur, S, T, Apk, in_dim);
    pack_w<<<nblk((long)Kc * D_OUT, TB), TB, 0, stream>>>(Wr[l], We[l], Bpk, in_dim);
    // h = relu(F @ Wc + br + deg*be)  -- WMMA, one wave per 16x128 strip
    float* hn = hbuf[l & 1];
    if (Kc == 192)
      gemm_node<192><<<(N_NODES / 16), 32, 0, stream>>>(Apk, Bpk, deg, be[l], br[l], hn);
    else
      gemm_node<320><<<(N_NODES / 16), 32, 0, stream>>>(Apk, Bpk, deg, be[l], br[l], hn);
    cur = hn;
    in_dim = D_OUT;
  }

  // global add pool -> [64, 128]
  zero_f32<<<nblk((long)N_GRAPHS * D_OUT, TB), TB, 0, stream>>>(outp, (long)N_GRAPHS * D_OUT);
  pool_add<<<nblk((long)N_NODES * D_OUT, TB), TB, 0, stream>>>(cur, batch, outp);
}
